// MllamaTextCrossAttention_738734375670
// MI455X (gfx1250) — matrix operand order repaired and checked by reading, not compiled, not run on hardware
//
#include <hip/hip_runtime.h>

// ---------- types ----------
typedef __attribute__((ext_vector_type(16))) __bf16          v16bf;
typedef __attribute__((ext_vector_type(2)))  __bf16          v2bf;
typedef __attribute__((ext_vector_type(8)))  float           v8f;
typedef __attribute__((ext_vector_type(16))) unsigned short  v16u;
typedef __attribute__((ext_vector_type(8)))  unsigned short  v8u;
typedef __attribute__((ext_vector_type(8)))  unsigned int    v8ui;

#define DEV __device__ __forceinline__

DEV v8u ld8(const unsigned short* p) { return *(const v8u*)p; }

DEV v16u cmb(v8u a, v8u b) {
  return __builtin_shufflevector(a, b, 0,1,2,3,4,5,6,7,8,9,10,11,12,13,14,15);
}

DEV v8f zero8() {
  v8f z;
  #pragma unroll
  for (int i = 0; i < 8; ++i) z[i] = 0.f;
  return z;
}

DEV v8f wmma_bf(v16u a, v16u b, v8f c) {
  // D(f32 16x16) = A(bf16 16x32) x B(bf16 32x16) + C
  return __builtin_amdgcn_wmma_f32_16x16x32_bf16(
      false, __builtin_bit_cast(v16bf, a),
      false, __builtin_bit_cast(v16bf, b),
      (short)0, c, false, false);
}

DEV unsigned short f2bf(float f) {  // RNE f32 -> bf16 (scalar fallback)
  unsigned u = __builtin_bit_cast(unsigned, f);
  u += 0x7fffu + ((u >> 16) & 1u);
  return (unsigned short)(u >> 16);
}

DEV unsigned pk2(float lo, float hi) {  // two f32 -> packed bf16x2
#if __has_builtin(__builtin_amdgcn_cvt_pk_bf16_f32)
  v2bf r = __builtin_amdgcn_cvt_pk_bf16_f32(lo, hi);
  return __builtin_bit_cast(unsigned, r);
#else
  return (unsigned)f2bf(lo) | ((unsigned)f2bf(hi) << 16);
#endif
}

// ---------- fp32 -> bf16 convert (packed 2-at-a-time) ----------
__global__ void k_cvt(const float* __restrict__ in, unsigned int* __restrict__ out, long n2) {
  long i = (long)blockIdx.x * blockDim.x + threadIdx.x;
  long s = (long)gridDim.x * blockDim.x;
  for (; i < n2; i += s) {
    float2 v = ((const float2*)in)[i];
    out[i] = pk2(v.x, v.y);
  }
}

// ---------- GEMM: C[M,N] = A[M,4096](bf16,rowmaj) * B[N,4096](bf16,rowmaj)^T ----------
// MODE 0: f32 store C[m*N+n].  MODE 1: bf16 transposed store Ct[n*ldT+m], zero for m>=M.
// One wave per 32x64 output tile. K fixed at 4096.
template <int MODE>
__global__ void __launch_bounds__(32, 1)
k_gemm(const unsigned short* __restrict__ A,
       const unsigned short* __restrict__ B,
       float* __restrict__ Cf, unsigned short* __restrict__ Ct,
       int M, int N, int ldT) {
  constexpr int K = 4096;
  const int lane = threadIdx.x;
  const int l16  = lane & 15;
  const bool hi  = lane >= 16;
  const int mBase = blockIdx.y * 32;
  const int nBase = blockIdx.x * 64;

  v8f c[2][4];
  #pragma unroll
  for (int i = 0; i < 2; ++i)
    #pragma unroll
    for (int j = 0; j < 4; ++j) c[i][j] = zero8();

  const int offA = hi ? 16 : 0;   // A k order (this kernel, same as B): lane<16 -> K 0..15 contiguous
  const int offB = hi ? 16 : 0;   // B k order (this kernel, same as A): lane<16 -> K 0..15 contiguous
  const int rowA0 = mBase + l16;
  const int rowA1 = mBase + 16 + l16;
  const bool okA0 = rowA0 < M, okA1 = rowA1 < M;
  const unsigned short* pA0 = A + (size_t)(okA0 ? rowA0 : 0) * K + offA;
  const unsigned short* pA1 = A + (size_t)(okA1 ? rowA1 : 0) * K + offA;
  const unsigned short* pB0 = B + (size_t)(nBase +  0 + l16) * K + offB;
  const unsigned short* pB1 = B + (size_t)(nBase + 16 + l16) * K + offB;
  const unsigned short* pB2 = B + (size_t)(nBase + 32 + l16) * K + offB;
  const unsigned short* pB3 = B + (size_t)(nBase + 48 + l16) * K + offB;

  for (int kb = 0; kb < K; kb += 32) {
    v16u a0 = cmb(ld8(pA0), ld8(pA0 + 8));
    v16u a1 = cmb(ld8(pA1), ld8(pA1 + 8));
    v16u b0 = cmb(ld8(pB0), ld8(pB0 + 8));
    v16u b1 = cmb(ld8(pB1), ld8(pB1 + 8));
    v16u b2 = cmb(ld8(pB2), ld8(pB2 + 8));
    v16u b3 = cmb(ld8(pB3), ld8(pB3 + 8));
    // prefetch next k-step (speculative; OOB past end is dropped harmlessly)
    __builtin_prefetch(pA0 + 32, 0, 3);
    __builtin_prefetch(pA1 + 32, 0, 3);
    __builtin_prefetch(pB0 + 32, 0, 3);
    __builtin_prefetch(pB1 + 32, 0, 3);
    __builtin_prefetch(pB2 + 32, 0, 3);
    __builtin_prefetch(pB3 + 32, 0, 3);
    if (!okA0) a0 = __builtin_bit_cast(v16u, zero8());
    if (!okA1) a1 = __builtin_bit_cast(v16u, zero8());
    c[0][0] = wmma_bf(a0, b0, c[0][0]);
    c[0][1] = wmma_bf(a0, b1, c[0][1]);
    c[0][2] = wmma_bf(a0, b2, c[0][2]);
    c[0][3] = wmma_bf(a0, b3, c[0][3]);
    c[1][0] = wmma_bf(a1, b0, c[1][0]);
    c[1][1] = wmma_bf(a1, b1, c[1][1]);
    c[1][2] = wmma_bf(a1, b2, c[1][2]);
    c[1][3] = wmma_bf(a1, b3, c[1][3]);
    pA0 += 32; pA1 += 32; pB0 += 32; pB1 += 32; pB2 += 32; pB3 += 32;
  }

  const int rOff = hi ? 8 : 0;
  #pragma unroll
  for (int mt = 0; mt < 2; ++mt) {
    #pragma unroll
    for (int nt = 0; nt < 4; ++nt) {
      int n = nBase + nt * 16 + l16;
      if (MODE == 0) {
        #pragma unroll
        for (int r = 0; r < 8; ++r) {
          int m = mBase + mt * 16 + rOff + r;
          if (m < M) Cf[(size_t)m * N + n] = c[mt][nt][r];
        }
      } else {
        #pragma unroll
        for (int r = 0; r < 8; r += 2) {
          int m = mBase + mt * 16 + rOff + r;   // even
          float v0 = (m     < M) ? c[mt][nt][r]     : 0.f;
          float v1 = (m + 1 < M) ? c[mt][nt][r + 1] : 0.f;
          *(unsigned*)(Ct + (size_t)n * ldT + m) = pk2(v0, v1);
        }
      }
    }
  }
}

// ---------- RMSNorm over head_dim=128, write bf16 head-major ----------
__global__ void k_rmsq(const float* __restrict__ X, const float* __restrict__ w,
                       unsigned short* __restrict__ out) {
  int q = blockIdx.x, h = blockIdx.y, lane = threadIdx.x;
  float4 x = *(const float4*)(X + (size_t)q * 4096 + h * 128 + lane * 4);
  float s = x.x * x.x + x.y * x.y + x.z * x.z + x.w * x.w;
  #pragma unroll
  for (int o = 16; o >= 1; o >>= 1) s += __shfl_xor(s, o, 32);
  float r = rsqrtf(s * (1.f / 128.f) + 1e-5f);
  float4 wv = *(const float4*)(w + lane * 4);
  unsigned* po = (unsigned*)(out + ((size_t)h * 1024 + q) * 128 + lane * 4);
  po[0] = pk2(wv.x * x.x * r, wv.y * x.y * r);
  po[1] = pk2(wv.z * x.z * r, wv.w * x.w * r);
}

__global__ void k_rmsk(const float* __restrict__ X, const float* __restrict__ w,
                       unsigned short* __restrict__ out) {
  constexpr int KV = 6404, KVP = 6464;
  int kv = blockIdx.x, hk = blockIdx.y, lane = threadIdx.x;
  unsigned* po = (unsigned*)(out + ((size_t)hk * KVP + kv) * 128 + lane * 4);
  if (kv >= KV) {                 // zero the pad rows so WMMA reads are clean
    po[0] = 0; po[1] = 0;
    return;
  }
  float4 x = *(const float4*)(X + (size_t)kv * 1024 + hk * 128 + lane * 4);
  float s = x.x * x.x + x.y * x.y + x.z * x.z + x.w * x.w;
  #pragma unroll
  for (int o = 16; o >= 1; o >>= 1) s += __shfl_xor(s, o, 32);
  float r = rsqrtf(s * (1.f / 128.f) + 1e-5f);
  float4 wv = *(const float4*)(w + lane * 4);
  po[0] = pk2(wv.x * x.x * r, wv.y * x.y * r);
  po[1] = pk2(wv.z * x.z * r, wv.w * x.w * r);
}

// ---------- Flash attention: 1 wave per (head, 16-query tile) ----------
// Computes S^T = K*Q^T so softmax over kv is in-lane; exp'd S^T C-layout is
// exactly the A-fragment layout of P for the P*V WMMAs (only bf16 packing needed).
// All dims compile-time so every fragment address is base + literal offset.
// launch_bounds(32,1): let the allocator use the full VGPR file (no spills).
__global__ void __launch_bounds__(32, 1)
k_flash(const unsigned short* __restrict__ Qn,
        const unsigned short* __restrict__ Kn,
        const unsigned short* __restrict__ Vt,
        const float* __restrict__ mask,
        unsigned short* __restrict__ ctx) {
  constexpr int QL = 1024, KV = 6404, KVP = 6464;
  const int lane = threadIdx.x;
  const int l16  = lane & 15;
  const bool hi  = lane >= 16;
  const int qBase = blockIdx.x * 16;
  const int h  = blockIdx.y;
  const int hk = h >> 2;                       // GQA: 4 query heads per KV head
  const unsigned short* Qh = Qn + (size_t)h  * QL  * 128;
  const unsigned short* Kh = Kn + (size_t)hk * KVP * 128;
  const unsigned short* Vh = Vt + (size_t)hk * 128 * KVP;  // Vt[h][d][kv]
  const float scale = 0.08838834764831845f;    // 1/sqrt(128)

  const int rOff = hi ? 8 : 0;
  const int offK = hi ? 16 : 0;
  const int offV = hi ? 8 : 0;
  const int offQ = hi ? 16 : 0;

  // Q as B-fragments of Q^T (lane = q column, contiguous d per lane)
  v16u bq[4];
  {
    const unsigned short* p0 = Qh + (size_t)(qBase + l16) * 128 + offQ;
    #pragma unroll
    for (int dc = 0; dc < 4; ++dc)
      bq[dc] = cmb(ld8(p0 + dc * 32), ld8(p0 + dc * 32 + 8));
  }

  v8f co[8];
  #pragma unroll
  for (int j = 0; j < 8; ++j) co[j] = zero8();
  float mold = -1e30f, lsum = 0.f;
  const int qAbs = qBase + l16;

  // single base pointers; all fragment addresses become literal offsets
  const unsigned short* kc = Kh + (size_t)l16 * 128 + offK;
  const unsigned short* vc = Vh + (size_t)l16 * KVP + offV;
  const float* mrow = mask + (size_t)qAbs * KV + rOff;

  auto chunk = [&](int kvB, bool guard) {
    // --- S^T tiles: rows = kv, cols = q ---
    v8f sT[4];
    #pragma unroll
    for (int t = 0; t < 4; ++t) {
      v8f acc = zero8();
      #pragma unroll
      for (int dc = 0; dc < 4; ++dc) {
        const unsigned short* p = kc + t * (16 * 128) + dc * 32;
        acc = wmma_bf(cmb(ld8(p), ld8(p + 8)), bq[dc], acc);
      }
      sT[t] = acc;
    }
    if (!guard) {   // prefetch next KV chunk of K
      #pragma unroll
      for (int t = 0; t < 4; ++t)
        __builtin_prefetch(kc + (64 + t * 16) * 128, 0, 3);
    }
    // --- scale + mask + online softmax stats ---
    float mloc = -1e30f;
    #pragma unroll
    for (int t = 0; t < 4; ++t) {
      #pragma unroll
      for (int r = 0; r < 8; ++r) {
        float s;
        if (!guard) {
          s = sT[t][r] * scale + mrow[kvB + t * 16 + r];
        } else {
          int kvA = kvB + t * 16 + rOff + r;
          s = (kvA < KV) ? (sT[t][r] * scale + mrow[kvB + t * 16 + r]) : -1e30f;
        }
        sT[t][r] = s;
        mloc = fmaxf(mloc, s);
      }
    }
    mloc = fmaxf(mloc, __shfl_xor(mloc, 16, 32));
    float mnew  = fmaxf(mold, mloc);
    float alpha = __expf(mold - mnew);
    float psum = 0.f;
    #pragma unroll
    for (int t = 0; t < 4; ++t)
      #pragma unroll
      for (int r = 0; r < 8; ++r) {
        float p = __expf(sT[t][r] - mnew);
        sT[t][r] = p;
        psum += p;
      }
    psum += __shfl_xor(psum, 16, 32);
    lsum = lsum * alpha + psum;
    mold = mnew;
    // --- pack P into A-fragments (in-register: S^T C-layout == P A-layout) ---
    v8ui p0, p1;
    #pragma unroll
    for (int i = 0; i < 4; ++i) {
      p0[i]     = pk2(sT[0][2 * i], sT[0][2 * i + 1]);
      p0[i + 4] = pk2(sT[1][2 * i], sT[1][2 * i + 1]);
      p1[i]     = pk2(sT[2][2 * i], sT[2][2 * i + 1]);
      p1[i + 4] = pk2(sT[3][2 * i], sT[3][2 * i + 1]);
    }
    v16u pa0 = __builtin_bit_cast(v16u, p0);
    v16u pa1 = __builtin_bit_cast(v16u, p1);
    // --- rescale O (per-q factors broadcast with bpermute) ---
    float fac[8];
    #pragma unroll
    for (int r = 0; r < 8; ++r) fac[r] = __shfl(alpha, rOff + r, 32);
    #pragma unroll
    for (int j = 0; j < 8; ++j)
      #pragma unroll
      for (int r = 0; r < 8; ++r) co[j][r] *= fac[r];
    // --- O += P * V (constant offsets off a single base) ---
    #pragma unroll
    for (int j = 0; j < 8; ++j) {
      const unsigned short* p = vc + (size_t)j * (16 * KVP);
      co[j] = wmma_bf(pa0, cmb(ld8(p),      ld8(p + 16)),     co[j]);
      co[j] = wmma_bf(pa1, cmb(ld8(p + 32), ld8(p + 48)),     co[j]);
    }
    if (!guard) {   // prefetch next KV chunk of V
      #pragma unroll
      for (int j = 0; j < 8; ++j)
        __builtin_prefetch(vc + (size_t)j * (16 * KVP) + 64, 0, 3);
    }
  };

  for (int kvB = 0; kvB < 6400; kvB += 64) {   // guard-free chunks (kv < KV)
    chunk(kvB, false);
    kc += 64 * 128;
    vc += 64;
  }
  chunk(6400, true);                            // tail chunk with masking

  // --- normalize and store context (bf16, [q][h*128+d]) ---
  float inv = 1.f / lsum;
  float fac[8];
  #pragma unroll
  for (int r = 0; r < 8; ++r) fac[r] = __shfl(inv, rOff + r, 32);
  #pragma unroll
  for (int j = 0; j < 8; ++j) {
    #pragma unroll
    for (int r = 0; r < 8; ++r) {
      int q   = qBase + rOff + r;
      int col = h * 128 + j * 16 + l16;
      ctx[(size_t)q * 4096 + col] = f2bf(co[j][r] * fac[r]);
    }
  }
}

// ---------- host ----------
extern "C" void kernel_launch(void* const* d_in, const int* in_sizes, int n_in,
                              void* d_out, int out_size, void* d_ws, size_t ws_size,
                              hipStream_t stream) {
  (void)in_sizes; (void)n_in; (void)out_size; (void)ws_size;
  constexpr int QL  = 1024;
  constexpr int KV  = 6404;
  constexpr int KVP = 6464;   // padded to multiple of 64
  constexpr int HID = 4096;
  constexpr int NKVH = 1024;  // 8 kv heads * 128

  const float* hs   = (const float*)d_in[0];
  const float* xs   = (const float*)d_in[1];
  const float* mask = (const float*)d_in[2];
  const float* wq   = (const float*)d_in[3];
  const float* wk   = (const float*)d_in[4];
  const float* wv   = (const float*)d_in[5];
  const float* wo   = (const float*)d_in[6];
  const float* qnw  = (const float*)d_in[7];
  const float* knw  = (const float*)d_in[8];
  float* out = (float*)d_out;

  unsigned char* wsb = (unsigned char*)d_ws;
  size_t off = 0;
  auto al = [&](size_t bytes) -> void* {
    void* p = wsb + off;
    off = (off + bytes + 255) & ~(size_t)255;
    return p;
  };
  unsigned short* hsb = (unsigned short*)al((size_t)QL * HID * 2);
  unsigned short* xsb = (unsigned short*)al((size_t)KV * HID * 2);
  unsigned short* wqb = (unsigned short*)al((size_t)HID * HID * 2);
  unsigned short* wkb = (unsigned short*)al((size_t)NKVH * HID * 2);
  unsigned short* wvb = (unsigned short*)al((size_t)NKVH * HID * 2);
  unsigned short* wob = (unsigned short*)al((size_t)HID * HID * 2);
  float*          qf  = (float*)al((size_t)QL * HID * 4);
  float*          kf  = (float*)al((size_t)KV * NKVH * 4);
  unsigned short* qn  = (unsigned short*)al((size_t)32 * QL * 128 * 2);
  unsigned short* kn  = (unsigned short*)al((size_t)8 * KVP * 128 * 2);
  unsigned short* vt  = (unsigned short*)al((size_t)8 * 128 * KVP * 2);
  unsigned short* ctx = (unsigned short*)al((size_t)QL * HID * 2);

  // 1) fp32 -> bf16 (packed)
  k_cvt<<<1024, 256, 0, stream>>>(hs, (unsigned int*)hsb, (long)QL * HID / 2);
  k_cvt<<<4096, 256, 0, stream>>>(xs, (unsigned int*)xsb, (long)KV * HID / 2);
  k_cvt<<<2048, 256, 0, stream>>>(wq, (unsigned int*)wqb, (long)HID * HID / 2);
  k_cvt<<<1024, 256, 0, stream>>>(wk, (unsigned int*)wkb, (long)NKVH * HID / 2);
  k_cvt<<<1024, 256, 0, stream>>>(wv, (unsigned int*)wvb, (long)NKVH * HID / 2);
  k_cvt<<<2048, 256, 0, stream>>>(wo, (unsigned int*)wob, (long)HID * HID / 2);

  // 2) projections (WMMA GEMMs)
  k_gemm<0><<<dim3(HID / 64, QL / 32), 32, 0, stream>>>(hsb, wqb, qf, nullptr, QL, HID, 0);
  k_gemm<0><<<dim3(NKVH / 64, (KV + 31) / 32), 32, 0, stream>>>(xsb, wkb, kf, nullptr, KV, NKVH, 0);
  k_gemm<1><<<dim3(NKVH / 64, KVP / 32), 32, 0, stream>>>(xsb, wvb, nullptr, vt, KV, NKVH, KVP);

  // 3) per-head RMSNorm -> bf16 head-major
  k_rmsq<<<dim3(QL, 32), 32, 0, stream>>>(qf, qnw, qn);
  k_rmsk<<<dim3(KVP, 8), 32, 0, stream>>>(kf, knw, kn);

  // 4) flash attention (WMMA)
  k_flash<<<dim3(QL / 16, 32), 32, 0, stream>>>(qn, kn, vt, mask, ctx);

  // 5) output projection (WMMA GEMM, f32 out)
  k_gemm<0><<<dim3(HID / 64, QL / 32), 32, 0, stream>>>(ctx, wob, out, nullptr, QL, HID, 0);
}
